// SparseConvolution_19963007992500
// MI455X (gfx1250) — compile-verified, matrix operand order checked by reading, not run on hardware
//
#include <hip/hip_runtime.h>

#define CIN   64
#define COUT  64
#define KVOL  27
#define MPAIR 50000
#define NVOX  100000

typedef __attribute__((ext_vector_type(16))) __bf16 v16bf;
typedef __attribute__((ext_vector_type(8)))  float  v8f;
typedef __attribute__((ext_vector_type(4)))  float  v4f;

// Non-returning fp32 global atomic add (native CDNA5 GLOBAL_ATOMIC_ADD_F32).
__device__ __forceinline__ void atomic_add_f32(float* p, float v) {
    asm volatile("global_atomic_add_f32 %0, %1, off" :: "v"(p), "v"(v) : "memory");
}

// out[i] = bias[i % COUT]  (rewrites the whole buffer every call -> replay safe)
__global__ __launch_bounds__(256) void init_out(float* __restrict__ out,
                                                const float* __restrict__ bias) {
    int i = blockIdx.x * blockDim.x + threadIdx.x;
    if (i < NVOX * COUT) out[i] = bias[i & (COUT - 1)];
}

__global__ __launch_bounds__(256) void spconv_wmma(
    const float* __restrict__ x,        // [N, CIN]
    const float* __restrict__ w,        // [K, CIN, COUT]
    const int*   __restrict__ in_map,   // [K, M]
    const int*   __restrict__ out_map,  // [K, M]
    float*       __restrict__ out)      // [N, COUT]
{
    const int k    = blockIdx.y;
    const int lane = threadIdx.x & 31;
    const int wid  = threadIdx.x >> 5;
    const int wavesTotal = gridDim.x * (blockDim.x >> 5);
    const int waveGlobal = blockIdx.x * (blockDim.x >> 5) + wid;
    const int half = lane >> 4;   // 0: K{0..7,16..23}, 1: K{8..15,24..31}
    const int l16  = lane & 15;   // A: row M, B: column N, C/D: column N

    // ---- B fragments for offset k: bf16 hi/lo split, loop-invariant ----
    const float* wk = w + (size_t)k * (CIN * COUT);
    v16bf Bhi[4][2], Blo[4][2];
#pragma unroll
    for (int n = 0; n < 4; ++n) {
        const int col = n * 16 + l16;
#pragma unroll
        for (int kb = 0; kb < 2; ++kb) {
            const int cb = kb * 32 + half * 8;
#pragma unroll
            for (int e = 0; e < 8; ++e) {
                float f0 = wk[(size_t)(cb + e) * COUT + col];
                float f1 = wk[(size_t)(cb + 16 + e) * COUT + col];
                __bf16 h0 = (__bf16)f0;
                __bf16 h1 = (__bf16)f1;
                Bhi[n][kb][e]     = h0;  Blo[n][kb][e]     = (__bf16)(f0 - (float)h0);
                Bhi[n][kb][8 + e] = h1;  Blo[n][kb][8 + e] = (__bf16)(f1 - (float)h1);
            }
        }
    }

    const int kbase  = k * MPAIR;
    const int ntiles = MPAIR / 16;   // 3125, exact -> no tail, EXEC stays full

    for (int t = waveGlobal; t < ntiles; t += wavesTotal) {
        const int mbase = t * 16;

        // ---- gather 16 rows of x, pack A (hi/lo bf16) per documented layout ----
        const int grow = in_map[kbase + mbase + l16];
        const float* xr = x + (size_t)grow * CIN;

        v16bf Ahi[2], Alo[2];
#pragma unroll
        for (int kb = 0; kb < 2; ++kb) {
            const int cb = kb * 32 + half * 8;
            v4f f0 = *(const v4f*)(xr + cb);
            v4f f1 = *(const v4f*)(xr + cb + 4);
            v4f f2 = *(const v4f*)(xr + cb + 16);
            v4f f3 = *(const v4f*)(xr + cb + 20);
#pragma unroll
            for (int e = 0; e < 4; ++e) {
                __bf16 h;
                h = (__bf16)f0[e]; Ahi[kb][e]      = h; Alo[kb][e]      = (__bf16)(f0[e] - (float)h);
                h = (__bf16)f1[e]; Ahi[kb][4 + e]  = h; Alo[kb][4 + e]  = (__bf16)(f1[e] - (float)h);
                h = (__bf16)f2[e]; Ahi[kb][8 + e]  = h; Alo[kb][8 + e]  = (__bf16)(f2[e] - (float)h);
                h = (__bf16)f3[e]; Ahi[kb][12 + e] = h; Alo[kb][12 + e] = (__bf16)(f3[e] - (float)h);
            }
        }

        // ---- 3-term compensated bf16 WMMA: hi*hi + lo*hi + hi*lo ----
        v8f acc[4] = {};
#pragma unroll
        for (int n = 0; n < 4; ++n) {
#pragma unroll
            for (int kb = 0; kb < 2; ++kb) {
                acc[n] = __builtin_amdgcn_wmma_f32_16x16x32_bf16(
                    false, Ahi[kb], false, Bhi[n][kb], (short)0, acc[n], false, false);
                acc[n] = __builtin_amdgcn_wmma_f32_16x16x32_bf16(
                    false, Alo[kb], false, Bhi[n][kb], (short)0, acc[n], false, false);
                acc[n] = __builtin_amdgcn_wmma_f32_16x16x32_bf16(
                    false, Ahi[kb], false, Blo[n][kb], (short)0, acc[n], false, false);
            }
        }

        // ---- scatter-add: VGPR j holds row (j + 8*half), lane l16 holds column ----
#pragma unroll
        for (int j = 0; j < 8; ++j) {
            const int r    = j + 8 * half;
            const int orow = out_map[kbase + mbase + r];
            float* op = out + (size_t)orow * COUT + l16;
#pragma unroll
            for (int n = 0; n < 4; ++n) {
                atomic_add_f32(op + n * 16, acc[n][j]);
            }
        }
    }
}

extern "C" void kernel_launch(void* const* d_in, const int* in_sizes, int n_in,
                              void* d_out, int out_size, void* d_ws, size_t ws_size,
                              hipStream_t stream) {
    const float* x       = (const float*)d_in[0];
    const float* w       = (const float*)d_in[1];
    const float* bias    = (const float*)d_in[2];
    const int*   in_map  = (const int*)d_in[3];
    const int*   out_map = (const int*)d_in[4];
    float*       out     = (float*)d_out;

    init_out<<<dim3((NVOX * COUT + 255) / 256), dim3(256), 0, stream>>>(out, bias);
    spconv_wmma<<<dim3(64, KVOL), dim3(256), 0, stream>>>(x, w, in_map, out_map, out);
}
